// DGNLB_12455405159000
// MI455X (gfx1250) — matrix operand order repaired and checked by reading, not compiled, not run on hardware
//
#include <hip/hip_runtime.h>

// ---------------------------------------------------------------------------
// DANet-style dual attention block for MI455X (gfx1250, wave32, WMMA).
// Dominant cost: gatt = softmax(att @ att_g), a 4096x4096x4096 GEMM (275 GFLOP)
// -> bf16 WMMA (v_wmma_f32_16x16x32_bf16) with f32 accumulate + f32 softmax.
// K=4096 GEMMs use async global->LDS staging of the B tile (ASYNCcnt path).
// ---------------------------------------------------------------------------

typedef __attribute__((ext_vector_type(16))) __bf16 v16bf;
typedef __attribute__((ext_vector_type(8)))  float  v8f;

#define NPIX 4096   // H*W
#define CCH  64
#define BB   2

__device__ __forceinline__ __bf16 f2bf(float x) {
    union { float f; unsigned u; } v; v.f = x;
    unsigned r = v.u + 0x7fffu + ((v.u >> 16) & 1u);   // round-to-nearest-even
    unsigned short h = (unsigned short)(r >> 16);
    union { unsigned short s; __bf16 b; } c; c.s = h;
    return c.b;
}

// ---- WMMA fragment loaders (contiguous-k order, same in A and B, wave32) -------------------
// A: 16x32 bf16 tile, row-major in memory with stride lda.
//   lane<16 : row=lane,    K = 0..15
//   lane>=16: row=lane-16, K = 16..31
__device__ __forceinline__ v16bf load_a_frag(const __bf16* __restrict__ A, int lda,
                                             int row0, int k0, int lane) {
    int r    = lane & 15;
    int koff = (lane >> 4) * 16;
    const __bf16* p = A + (size_t)(row0 + r) * lda + k0 + koff;
    union { uint4 u[2]; v16bf v; } f;
    f.u[0] = *reinterpret_cast<const uint4*>(p);
    f.u[1] = *reinterpret_cast<const uint4*>(p + 8);
    return f.v;
}

// B: 32x16 bf16 tile; operand supplied column-contiguous: memory holds
// Bt[col][k] with stride ldb (i.e. B transposed, K fastest).
//   lane<16 : col=lane,    K = 0..15
//   lane>=16: col=lane-16, K = 16..31
__device__ __forceinline__ v16bf load_b_frag(const __bf16* __restrict__ Bt, int ldb,
                                             int col0, int k0, int lane) {
    int c    = lane & 15;
    int koff = (lane >> 4) * 16;
    const __bf16* p = Bt + (size_t)(col0 + c) * ldb + k0 + koff;
    union { uint4 u[2]; v16bf v; } f;
    f.u[0] = *reinterpret_cast<const uint4*>(p);
    f.u[1] = *reinterpret_cast<const uint4*>(p + 8);
    return f.v;
}

// ---- direct-global bf16 GEMM (used for K=64 score GEMMs) ------------------
// D[M,N] = A[M,K] * Bt[N,K]^T, f32 accumulate. One wave per 16x64 strip.
__global__ void gemm_bf16_wmma(const __bf16* __restrict__ A, int lda,
                               const __bf16* __restrict__ Bt, int ldb,
                               float* __restrict__ Cout, int ldc, int K,
                               int mode, const float* __restrict__ resid,
                               const float* __restrict__ gamma) {
    int lane = threadIdx.x & 31;
    int row0 = blockIdx.y * 16;
    int col0 = blockIdx.x * 64;
    v8f acc[4] = {{}, {}, {}, {}};
    for (int k0 = 0; k0 < K; k0 += 32) {
        v16bf a = load_a_frag(A, lda, row0, k0, lane);
#pragma unroll
        for (int t = 0; t < 4; ++t) {
            v16bf b = load_b_frag(Bt, ldb, col0 + t * 16, k0, lane);
            acc[t] = __builtin_amdgcn_wmma_f32_16x16x32_bf16(
                false, a, false, b, (short)0, acc[t], false, false);
        }
    }
    int c     = lane & 15;
    int rbase = (lane >> 4) * 8;
    float gm  = (mode == 1) ? gamma[0] : 0.0f;
#pragma unroll
    for (int t = 0; t < 4; ++t) {
        int col = col0 + t * 16 + c;
#pragma unroll
        for (int i = 0; i < 8; ++i) {
            size_t idx = (size_t)(row0 + rbase + i) * ldc + col;
            float  val = acc[t][i];
            if (mode == 1) val = gm * val + resid[idx];
            Cout[idx] = val;
        }
    }
}

// ---- LDS-staged bf16 GEMM for K=4096 --------------------------------------
// Block = 128 threads = 4 waves, 64x64 output tile. Per K-step the block
// async-copies the 64x32 B tile (4 KB) to LDS (global_load_async_to_lds_b128,
// ASYNCcnt), then all 4 waves consume it via LDS b128 reads -> 4x less B
// traffic to L2 than the direct kernel. A-frags stream from global with
// prefetch. mode 0: f32 store; mode 1: gamma*acc + resid.
__global__ void gemm_bf16_wmma_lds(const __bf16* __restrict__ A, int lda,
                                   const __bf16* __restrict__ Bt, int ldb,
                                   float* __restrict__ Cout, int ldc, int K,
                                   int mode, const float* __restrict__ resid,
                                   const float* __restrict__ gamma) {
    __shared__ __bf16 bt[64 * 32];   // [colLocal][k], k fastest, 4 KB
    int tid  = threadIdx.x;
    int wave = tid >> 5;
    int lane = tid & 31;
    int row0 = blockIdx.y * 64 + wave * 16;
    int col0 = blockIdx.x * 64;

    // per-thread cooperative-copy coordinates: 256 16B chunks, 2 per thread
    int colL0 = tid >> 2;
    int ko0   = (tid & 3) * 8;
    int colL1 = (tid + 128) >> 2;
    int ko1   = ((tid + 128) & 3) * 8;
    unsigned ldsOff0 = (unsigned)(size_t)(void*)&bt[colL0 * 32 + ko0];
    unsigned ldsOff1 = (unsigned)(size_t)(void*)&bt[colL1 * 32 + ko1];

    v8f acc[4] = {{}, {}, {}, {}};
    for (int k0 = 0; k0 < K; k0 += 32) {
        // async global -> LDS copy of the 64x32 B tile
        const __bf16* src0 = Bt + (size_t)(col0 + colL0) * ldb + k0 + ko0;
        const __bf16* src1 = Bt + (size_t)(col0 + colL1) * ldb + k0 + ko1;
        asm volatile("global_load_async_to_lds_b128 %0, %1, off"
                     :: "v"(ldsOff0), "v"((unsigned long long)(size_t)src0) : "memory");
        asm volatile("global_load_async_to_lds_b128 %0, %1, off"
                     :: "v"(ldsOff1), "v"((unsigned long long)(size_t)src1) : "memory");

        // A fragment direct from global while the copy is in flight
        v16bf a = load_a_frag(A, lda, row0, k0, lane);
        if (k0 + 64 < K)   // cover latency of the next A line
            __builtin_prefetch(A + (size_t)(row0 + lane) * lda + k0 + 64, 0, 1);

        asm volatile("s_wait_asynccnt 0x0" ::: "memory");
        __syncthreads();

        int cL   = lane & 15;
        int koff = (lane >> 4) * 16;
#pragma unroll
        for (int t = 0; t < 4; ++t) {
            const uint4* p =
                reinterpret_cast<const uint4*>(&bt[(t * 16 + cL) * 32 + koff]);
            union { uint4 u[2]; v16bf v; } f;
            f.u[0] = p[0];
            f.u[1] = p[1];
            acc[t] = __builtin_amdgcn_wmma_f32_16x16x32_bf16(
                false, a, false, f.v, (short)0, acc[t], false, false);
        }
        __syncthreads();   // protect bt before next iteration's overwrite
    }
    int c     = lane & 15;
    int rbase = (lane >> 4) * 8;
    float gm  = (mode == 1) ? gamma[0] : 0.0f;
#pragma unroll
    for (int t = 0; t < 4; ++t) {
        int col = col0 + t * 16 + c;
#pragma unroll
        for (int i = 0; i < 8; ++i) {
            size_t idx = (size_t)(row0 + rbase + i) * ldc + col;
            float  val = acc[t][i];
            if (mode == 1) val = gm * val + resid[idx];
            Cout[idx] = val;
        }
    }
}

// ---- fused 1x1 projections: q,k,v,qg,kg ----------------------------------
// qT/kT/qgT/kgT stored [b][n][c] bf16 (K-contiguous for WMMA); vB [b][c][n].
__global__ void proj5_kernel(const float* __restrict__ x, const float* __restrict__ g,
                             const float* qw, const float* qb,
                             const float* kw, const float* kb,
                             const float* vw, const float* vb,
                             const float* qgw, const float* qgb,
                             const float* kgw, const float* kgb,
                             __bf16* qT, __bf16* kT, __bf16* qgT, __bf16* kgT,
                             __bf16* vB) {
    int idx = blockIdx.x * blockDim.x + threadIdx.x;   // [b][n][c], c fastest
    int c = idx & 63, n = (idx >> 6) & (NPIX - 1), b = idx >> 18;
    const float* xp = x + ((size_t)b * CCH) * NPIX + n;
    const float* gp = g + ((size_t)b * CCH) * NPIX + n;
    float sq = qb[c], sk = kb[c], sv = vb[c], sqg = qgb[c], skg = kgb[c];
    for (int i = 0; i < CCH; ++i) {
        float xv = xp[(size_t)i * NPIX];
        float gv = gp[(size_t)i * NPIX];
        sq  += qw [c * CCH + i] * xv;
        sk  += kw [c * CCH + i] * xv;
        sv  += vw [c * CCH + i] * xv;
        sqg += qgw[c * CCH + i] * gv;
        skg += kgw[c * CCH + i] * gv;
    }
    size_t onc = ((size_t)b * NPIX + n) * CCH + c;
    qT[onc]  = f2bf(sq);
    kT[onc]  = f2bf(sk);
    qgT[onc] = f2bf(sqg);
    kgT[onc] = f2bf(skg);
    vB[((size_t)b * CCH + c) * NPIX + n] = f2bf(sv);
}

// ---- row softmax f32 -> bf16 (optionally transposed store) ----------------
__global__ void softmax_row_bf16(const float* __restrict__ S, __bf16* __restrict__ out,
                                 int cols, int transpose, int ld_out) {
    int row = blockIdx.x, tid = threadIdx.x;
    const float* sp = S + (size_t)row * cols;
    __shared__ float red[256];
    float m = -1e30f;
    for (int j = tid; j < cols; j += 256) m = fmaxf(m, sp[j]);
    red[tid] = m; __syncthreads();
    for (int s = 128; s > 0; s >>= 1) {
        if (tid < s) red[tid] = fmaxf(red[tid], red[tid + s]);
        __syncthreads();
    }
    m = red[0]; __syncthreads();
    float sum = 0.f;
    for (int j = tid; j < cols; j += 256) sum += __expf(sp[j] - m);
    red[tid] = sum; __syncthreads();
    for (int s = 128; s > 0; s >>= 1) {
        if (tid < s) red[tid] += red[tid + s];
        __syncthreads();
    }
    float inv = 1.0f / red[0];
    for (int j = tid; j < cols; j += 256) {
        float v = __expf(sp[j] - m) * inv;
        if (transpose) out[(size_t)j * ld_out + row] = f2bf(v);
        else           out[(size_t)row * ld_out + j] = f2bf(v);
    }
}

// ---- small 64-wide row softmax, f32 out (channel attention) ---------------
__global__ void softmax_row64_f32(const float* __restrict__ S, float* __restrict__ out) {
    int r = threadIdx.x;   // 64 rows, one thread each
    const float* sp = S + r * 64;
    float m = -1e30f;
    for (int j = 0; j < 64; ++j) m = fmaxf(m, sp[j]);
    float s = 0.f;
    for (int j = 0; j < 64; ++j) s += __expf(sp[j] - m);
    float inv = 1.0f / s;
    for (int j = 0; j < 64; ++j) out[r * 64 + j] = __expf(sp[j] - m) * inv;
}

// ---- 3x3 conv + BN + PReLU ------------------------------------------------
__global__ void conv3x3_bn_prelu(const float* __restrict__ in, const float* __restrict__ w,
                                 const float* bias, const float* bs, const float* bb,
                                 const float* bm, const float* bv, const float* alpha,
                                 float* __restrict__ out) {
    int idx = blockIdx.x * blockDim.x + threadIdx.x;
    int xw = idx & 63, y = (idx >> 6) & 63, co = (idx >> 12) & 63, b = idx >> 18;
    const float* ib = in + (size_t)b * CCH * NPIX;
    const float* wb = w + (size_t)co * CCH * 9;
    float s = bias[co];
    for (int ci = 0; ci < CCH; ++ci) {
        const float* ic = ib + (size_t)ci * NPIX;
        const float* wc = wb + ci * 9;
#pragma unroll
        for (int dy = 0; dy < 3; ++dy) {
            int yy = y + dy - 1;
            if ((unsigned)yy < 64u) {
#pragma unroll
                for (int dx = 0; dx < 3; ++dx) {
                    int xx = xw + dx - 1;
                    if ((unsigned)xx < 64u) s += wc[dy * 3 + dx] * ic[yy * 64 + xx];
                }
            }
        }
    }
    float inv = bs[co] * rsqrtf(bv[co] + 1e-5f);
    float val = s * inv + (bb[co] - bm[co] * inv);
    out[(size_t)idx] = val > 0.f ? val : alpha[0] * val;
}

// ---- 1x1 conv + BN + PReLU (optional bf16 shadow copy) --------------------
__global__ void conv1x1_bn_prelu(const float* __restrict__ in, const float* __restrict__ w,
                                 const float* bias, const float* bs, const float* bb,
                                 const float* bm, const float* bv, const float* alpha,
                                 float* __restrict__ out, __bf16* __restrict__ out_bf) {
    int idx = blockIdx.x * blockDim.x + threadIdx.x;
    int n = idx & (NPIX - 1), c = (idx >> 12) & 63, b = idx >> 18;
    const float* ip = in + ((size_t)b * CCH) * NPIX + n;
    float s = bias[c];
    for (int i = 0; i < CCH; ++i) s += w[c * CCH + i] * ip[(size_t)i * NPIX];
    float inv = bs[c] * rsqrtf(bv[c] + 1e-5f);
    float val = s * inv + (bb[c] - bm[c] * inv);
    val = val > 0.f ? val : alpha[0] * val;
    out[(size_t)idx] = val;
    if (out_bf) out_bf[(size_t)idx] = f2bf(val);
}

// ---- SE gate --------------------------------------------------------------
__global__ void se_reduce(const float* __restrict__ g, float* __restrict__ gy) {
    int bc = blockIdx.x, tid = threadIdx.x;
    const float* p = g + (size_t)bc * NPIX;
    __shared__ float red[256];
    float s = 0.f;
    for (int j = tid; j < NPIX; j += 256) s += p[j];
    red[tid] = s; __syncthreads();
    for (int st = 128; st > 0; st >>= 1) {
        if (tid < st) red[tid] += red[tid + st];
        __syncthreads();
    }
    if (tid == 0) gy[bc] = red[0] * (1.0f / NPIX);
}

__global__ void se_fc(const float* __restrict__ gy, const float* __restrict__ fc1,
                      const float* __restrict__ fc2, float* __restrict__ gate) {
    int b = blockIdx.x, t = threadIdx.x;   // 64 threads
    __shared__ float r[32];
    __shared__ float gys[64];
    gys[t] = gy[b * 64 + t]; __syncthreads();
    if (t < 32) {
        float s = 0.f;
        for (int c = 0; c < 64; ++c) s += fc1[t * 64 + c] * gys[c];
        r[t] = fmaxf(s, 0.f);
    }
    __syncthreads();
    float s = 0.f;
    for (int j = 0; j < 32; ++j) s += fc2[t * 32 + j] * r[j];
    gate[b * 64 + t] = 1.0f / (1.0f + __expf(-s));
}

__global__ void gq_gate_bf16(const float* __restrict__ g, const float* __restrict__ gate,
                             __bf16* __restrict__ out) {
    int idx = blockIdx.x * blockDim.x + threadIdx.x;
    int c = (idx >> 12) & 63, b = idx >> 18;
    out[(size_t)idx] = f2bf(g[(size_t)idx] * gate[b * 64 + c]);
}

// ---- CAM: ge = att_c@att_cg ; gatt_c = softmax(rowmax(ge) - ge) -----------
__global__ void cam_combine(const float* __restrict__ attc, const float* __restrict__ attcg,
                            float* __restrict__ gattc) {
    int b = blockIdx.x;
    __shared__ float ge[64 * 64];
    const float* ac  = attc  + (size_t)b * 4096;
    const float* acg = attcg + (size_t)b * 4096;
    for (int t = threadIdx.x; t < 4096; t += 256) {
        int cc = t >> 6, d = t & 63;
        float s = 0.f;
        for (int k = 0; k < 64; ++k) s += ac[cc * 64 + k] * acg[k * 64 + d];
        ge[t] = s;
    }
    __syncthreads();
    if (threadIdx.x < 64) {
        int cc = threadIdx.x;
        float mx = -1e30f;
        for (int d = 0; d < 64; ++d) mx = fmaxf(mx, ge[cc * 64 + d]);
        float m2 = -1e30f;
        for (int d = 0; d < 64; ++d) m2 = fmaxf(m2, mx - ge[cc * 64 + d]);
        float sum = 0.f;
        for (int d = 0; d < 64; ++d) sum += __expf(mx - ge[cc * 64 + d] - m2);
        float inv = 1.0f / sum;
        for (int d = 0; d < 64; ++d)
            gattc[(size_t)b * 4096 + cc * 64 + d] = __expf(mx - ge[cc * 64 + d] - m2) * inv;
    }
}

__global__ void cam_apply(const float* __restrict__ gattc, const float* __restrict__ pam2,
                          const float* __restrict__ gamma, float* __restrict__ cam) {
    int idx = blockIdx.x * blockDim.x + threadIdx.x;
    int n = idx & (NPIX - 1), c = (idx >> 12) & 63, b = idx >> 18;
    const float* gr = gattc + (size_t)b * 4096 + c * 64;
    const float* pp = pam2 + ((size_t)b * CCH) * NPIX + n;
    float s = 0.f;
    for (int d = 0; d < 64; ++d) s += gr[d] * pp[(size_t)d * NPIX];
    cam[(size_t)idx] = gamma[0] * s + pam2[(size_t)idx];
}

// ---------------------------------------------------------------------------
extern "C" void kernel_launch(void* const* d_in, const int* in_sizes, int n_in,
                              void* d_out, int out_size, void* d_ws, size_t ws_size,
                              hipStream_t stream) {
    (void)in_sizes; (void)n_in; (void)out_size; (void)ws_size;
    const float* x   = (const float*)d_in[0];
    const float* g   = (const float*)d_in[1];
    const float* qw  = (const float*)d_in[2],  *qb2  = (const float*)d_in[3];
    const float* kw  = (const float*)d_in[4],  *kb2  = (const float*)d_in[5];
    const float* vw  = (const float*)d_in[6],  *vb2  = (const float*)d_in[7];
    const float* qgw = (const float*)d_in[8],  *qgb2 = (const float*)d_in[9];
    const float* kgw = (const float*)d_in[10], *kgb2 = (const float*)d_in[11];
    const float* gamma_p = (const float*)d_in[12];
    const float* fc1 = (const float*)d_in[13], *fc2 = (const float*)d_in[14];
    const float* gamma_c = (const float*)d_in[15];
    const float* pc1w = (const float*)d_in[16], *pc1b = (const float*)d_in[17];
    const float* pb1s = (const float*)d_in[18], *pb1b = (const float*)d_in[19];
    const float* pb1m = (const float*)d_in[20], *pb1v = (const float*)d_in[21];
    const float* ppr1 = (const float*)d_in[22];
    const float* pc2w = (const float*)d_in[23], *pc2b = (const float*)d_in[24];
    const float* pb2s = (const float*)d_in[25], *pb2b = (const float*)d_in[26];
    const float* pb2m = (const float*)d_in[27], *pb2v = (const float*)d_in[28];
    const float* ppr2 = (const float*)d_in[29];
    const float* cc1w = (const float*)d_in[30], *cc1b = (const float*)d_in[31];
    const float* cb1s = (const float*)d_in[32], *cb1b = (const float*)d_in[33];
    const float* cb1m = (const float*)d_in[34], *cb1v = (const float*)d_in[35];
    const float* cpr1 = (const float*)d_in[36];
    const float* cc2w = (const float*)d_in[37], *cc2b = (const float*)d_in[38];
    const float* cb2s = (const float*)d_in[39], *cb2b = (const float*)d_in[40];
    const float* cb2m = (const float*)d_in[41], *cb2v = (const float*)d_in[42];
    const float* cpr2 = (const float*)d_in[43];
    const float* fcw  = (const float*)d_in[44], *fcb  = (const float*)d_in[45];
    const float* fbs  = (const float*)d_in[46], *fbb  = (const float*)d_in[47];
    const float* fbm  = (const float*)d_in[48], *fbv  = (const float*)d_in[49];
    const float* fpr  = (const float*)d_in[50];

    const size_t N = NPIX, C = CCH, NN = N * N;
    char* cur = (char*)d_ws;
    auto alloc = [&](size_t bytes) -> void* {
        void* p = cur;
        cur += (bytes + 255) & ~(size_t)255;
        return p;
    };
    __bf16* qT    = (__bf16*)alloc(BB * N * C * 2);
    __bf16* kT    = (__bf16*)alloc(BB * N * C * 2);
    __bf16* qgT   = (__bf16*)alloc(BB * N * C * 2);
    __bf16* kgT   = (__bf16*)alloc(BB * N * C * 2);
    __bf16* vB    = (__bf16*)alloc(BB * C * N * 2);
    __bf16* attA  = (__bf16*)alloc(BB * NN * 2);     // softmax(q^T k), [n][k]
    __bf16* attGT = (__bf16*)alloc(BB * NN * 2);     // softmax(qg^T kg) transposed [m][k]
    __bf16* gatt  = (__bf16*)alloc(BB * NN * 2);     // softmax(attA @ attG), [n][m]
    float*  Sf    = (float*)alloc(NN * 4);           // per-batch f32 GEMM scratch
    float*  pam   = (float*)alloc(BB * C * N * 4);
    float*  tbuf  = (float*)alloc(BB * C * N * 4);
    float*  pam2  = (float*)alloc(BB * C * N * 4);
    float*  cam   = (float*)alloc(BB * C * N * 4);
    float*  cam2  = (float*)alloc(BB * C * N * 4);
    __bf16* pam2b = (__bf16*)alloc(BB * C * N * 2);
    __bf16* gqb   = (__bf16*)alloc(BB * C * N * 2);
    float*  gym   = (float*)alloc(BB * C * 4);
    float*  gate  = (float*)alloc(BB * C * 4);
    float*  attc  = (float*)alloc(BB * 64 * 64 * 4);
    float*  attcg = (float*)alloc(BB * 64 * 64 * 4);
    float*  gattc = (float*)alloc(BB * 64 * 64 * 4);
    float*  smallS = (float*)alloc(64 * 64 * 4);

    const int TOT = BB * (int)C * (int)N;   // 524288
    const int BLK = TOT / 256;              // 2048

    // 1) fused q/k/v/qg/kg projections (bf16 outputs, WMMA-ready layouts)
    proj5_kernel<<<BLK, 256, 0, stream>>>(x, g, qw, qb2, kw, kb2, vw, vb2,
                                          qgw, qgb2, kgw, kgb2, qT, kT, qgT, kgT, vB);

    // 2) per-batch: att = softmax(q^T k), att_g^T = softmax(qg^T kg)^T  (K=64)
    for (int b = 0; b < BB; ++b) {
        size_t oNC = (size_t)b * N * C, oNN = (size_t)b * NN;
        gemm_bf16_wmma<<<dim3(N / 64, N / 16), 32, 0, stream>>>(
            qT + oNC, (int)C, kT + oNC, (int)C, Sf, (int)N, (int)C, 0, nullptr, nullptr);
        softmax_row_bf16<<<(int)N, 256, 0, stream>>>(Sf, attA + oNN, (int)N, 0, (int)N);
        gemm_bf16_wmma<<<dim3(N / 64, N / 16), 32, 0, stream>>>(
            qgT + oNC, (int)C, kgT + oNC, (int)C, Sf, (int)N, (int)C, 0, nullptr, nullptr);
        softmax_row_bf16<<<(int)N, 256, 0, stream>>>(Sf, attGT + oNN, (int)N, 1, (int)N);
    }

    // 3) the big one: gatt = softmax(attA @ attG)  (4096^3, LDS-staged WMMA)
    //    then pam = gamma_p * (v @ gatt^T) + x   (residual fused in epilogue)
    for (int b = 0; b < BB; ++b) {
        size_t oNN = (size_t)b * NN, oCN = (size_t)b * C * N;
        gemm_bf16_wmma_lds<<<dim3(N / 64, N / 64), 128, 0, stream>>>(
            attA + oNN, (int)N, attGT + oNN, (int)N, Sf, (int)N, (int)N, 0, nullptr, nullptr);
        softmax_row_bf16<<<(int)N, 256, 0, stream>>>(Sf, gatt + oNN, (int)N, 0, (int)N);
        gemm_bf16_wmma_lds<<<dim3(N / 64, C / 64), 128, 0, stream>>>(
            vB + oCN, (int)N, gatt + oNN, (int)N, pam + oCN, (int)N, (int)N,
            1, x + oCN, gamma_p);
    }

    // 4) pconv head
    conv3x3_bn_prelu<<<BLK, 256, 0, stream>>>(pam, pc1w, pc1b, pb1s, pb1b, pb1m, pb1v, ppr1, tbuf);
    conv1x1_bn_prelu<<<BLK, 256, 0, stream>>>(tbuf, pc2w, pc2b, pb2s, pb2b, pb2m, pb2v, ppr2,
                                              pam2, pam2b);

    // 5) SE gate on guide, gq = g * sigmoid(fc2(relu(fc1(mean(g)))))
    se_reduce<<<BB * (int)C, 256, 0, stream>>>(g, gym);
    se_fc<<<BB, 64, 0, stream>>>(gym, fc1, fc2, gate);
    gq_gate_bf16<<<BLK, 256, 0, stream>>>(g, gate, gqb);

    // 6) channel attentions (64x64, K=4096) via LDS-staged WMMA, f32 softmax
    for (int b = 0; b < BB; ++b) {
        size_t oCN = (size_t)b * C * N;
        gemm_bf16_wmma_lds<<<dim3(1, 1), 128, 0, stream>>>(
            pam2b + oCN, (int)N, pam2b + oCN, (int)N, smallS, 64, (int)N, 0, nullptr, nullptr);
        softmax_row64_f32<<<1, 64, 0, stream>>>(smallS, attc + (size_t)b * 4096);
        gemm_bf16_wmma_lds<<<dim3(1, 1), 128, 0, stream>>>(
            gqb + oCN, (int)N, gqb + oCN, (int)N, smallS, 64, (int)N, 0, nullptr, nullptr);
        softmax_row64_f32<<<1, 64, 0, stream>>>(smallS, attcg + (size_t)b * 4096);
    }
    cam_combine<<<BB, 256, 0, stream>>>(attc, attcg, gattc);
    cam_apply<<<BLK, 256, 0, stream>>>(gattc, pam2, gamma_c, cam);

    // 7) cconv + fconv heads
    conv3x3_bn_prelu<<<BLK, 256, 0, stream>>>(cam, cc1w, cc1b, cb1s, cb1b, cb1m, cb1v, cpr1, tbuf);
    conv1x1_bn_prelu<<<BLK, 256, 0, stream>>>(tbuf, cc2w, cc2b, cb2s, cb2b, cb2m, cb2v, cpr2,
                                              cam2, nullptr);
    conv1x1_bn_prelu<<<BLK, 256, 0, stream>>>(cam2, fcw, fcb, fbs, fbb, fbm, fbv, fpr,
                                              (float*)d_out, nullptr);
}
